// TransformerBlock_43276090474711
// MI455X (gfx1250) — compile-verified, matrix operand order checked by reading, not run on hardware
//
#include <hip/hip_runtime.h>
#include <math.h>

// ---------------- problem constants (match reference) ----------------
#define B_    2
#define S_    1024
#define D_    1024
#define H_    16
#define HD_   64
#define E_    8
#define HID_  4096
#define T_    (B_ * S_)          // 2048 tokens
#define MAXROWS 4608             // T_*TOPK (=4096) + 8 experts * 63 pad, 64-aligned

typedef __attribute__((ext_vector_type(16))) __bf16 v16bf;
typedef __attribute__((ext_vector_type(8)))  float  v8f;

struct __align__(16) U4 { unsigned x, y, z, w; };

static __device__ __forceinline__ __bf16 f2bf(float x) { return (__bf16)x; }

// ---- CDNA5 async global->LDS copy (ASYNCcnt path), with safe fallback ------
#if defined(__gfx1250__) && __has_builtin(__builtin_amdgcn_global_load_async_to_lds_b128)
#define HAVE_ASYNC_LDS 1
#else
#define HAVE_ASYNC_LDS 0
#endif

#if HAVE_ASYNC_LDS
typedef int v4i __attribute__((vector_size(16)));
typedef __attribute__((address_space(1))) v4i* g_v4i_ptr;
typedef __attribute__((address_space(3))) v4i* l_v4i_ptr;
#endif

static __device__ __forceinline__ void async_cp16(const __bf16* g, __bf16* l) {
#if HAVE_ASYNC_LDS
  __builtin_amdgcn_global_load_async_to_lds_b128(
      (g_v4i_ptr)(void*)g, (l_v4i_ptr)(void*)l, 0, 0);
#else
  *(U4*)l = *(const U4*)g;          // synchronous staging fallback
#endif
}

static __device__ __forceinline__ void wait_async_all() {
#if HAVE_ASYNC_LDS
#if __has_builtin(__builtin_amdgcn_s_wait_asynccnt)
  __builtin_amdgcn_s_wait_asynccnt(0);
#else
  asm volatile("s_wait_asynccnt 0" ::: "memory");
#endif
#endif
}

// Load a 16x32 bf16 WMMA A/B fragment from a row-major tile (global or LDS).
// Caller passes per-lane row pointer (row = lane&15) at the K-tile start.
// ISA 7.12.2: lanes 0-15 hold K {0..7,16..23}, lanes 16-31 hold K {8..15,24..31}
// -> 16B granules {g, 2+g}, g = (lane>=16).
static __device__ __forceinline__ v16bf load_frag(const __bf16* rowPtr, int lane) {
  const U4* p = (const U4*)rowPtr;
  int g = (lane & 16) ? 1 : 0;
  union { v16bf v; U4 q[2]; } f;
  f.q[0] = p[g];
  f.q[1] = p[2 + g];
  return f.v;
}

static __device__ __forceinline__ v8f wmma_bf16(v16bf a, v16bf b, v8f c) {
  return __builtin_amdgcn_wmma_f32_16x16x32_bf16(false, a, false, b, (short)0, c,
                                                 false, false);
}

// ---------------- weight convert + transpose: W[K][N] f32 -> Wt[N][K] bf16 ---
__global__ void transposeW(const float* __restrict__ W, __bf16* __restrict__ Wt,
                           int K, int N) {
  __shared__ float tile[32][33];
  size_t eoff = (size_t)blockIdx.z * K * N;
  const float* Wp = W + eoff;
  __bf16* Wtp = Wt + eoff;
  int kt = blockIdx.x * 32, nt = blockIdx.y * 32;
  int tx = threadIdx.x;
  for (int j = threadIdx.y; j < 32; j += 8)
    tile[j][tx] = Wp[(size_t)(kt + j) * N + nt + tx];
  __syncthreads();
  for (int j = threadIdx.y; j < 32; j += 8)
    Wtp[(size_t)(nt + j) * K + kt + tx] = f2bf(tile[tx][j]);
}

// ---------------- rmsnorm ----------------------------------------------------
__global__ void rmsnorm_kernel(const float* __restrict__ x, const float* __restrict__ w,
                               __bf16* __restrict__ outB, float* __restrict__ outF) {
  __shared__ float red[256];
  int t = blockIdx.x, tid = threadIdx.x;
  const float* xr = x + (size_t)t * D_;
  float s = 0.f;
  for (int d = tid; d < D_; d += 256) { float v = xr[d]; s += v * v; }
  red[tid] = s;
  __syncthreads();
  for (int st = 128; st > 0; st >>= 1) {
    if (tid < st) red[tid] += red[tid + st];
    __syncthreads();
  }
  float inv = rsqrtf(red[0] * (1.f / D_) + 1e-6f);
  for (int d = tid; d < D_; d += 256) {
    float v = xr[d] * inv * w[d];
    outB[(size_t)t * D_ + d] = f2bf(v);
    if (outF) outF[(size_t)t * D_ + d] = v;
  }
}

// ---------------- unified LDS-tiled WMMA GEMM --------------------------------
// 128 threads (4 waves) per block, 64(M)x64(N) macro tile, K step 32.
// A,B tiles double-buffered in LDS via async global->LDS; each wave computes a
// 16x64 strip (4 WMMA per K step, 16 per block per K step).
__global__ __launch_bounds__(128) void gemm_tiled(
    const __bf16* __restrict__ A, const __bf16* __restrict__ WtAll,
    const float* __restrict__ biasAll, const float* __restrict__ residual,
    float* __restrict__ outF, __bf16* __restrict__ outB, int K, int N,
    const int* __restrict__ off, const int* __restrict__ pcnt,
    const int* __restrict__ ptot, const int* __restrict__ assign_tok,
    int gather, int gelu) {
  __shared__ __align__(16) __bf16 sA[2][64 * 32];
  __shared__ __align__(16) __bf16 sB[2][64 * 32];
  int tid = threadIdx.x;
  int row0 = blockIdx.y * 64;
  if (ptot && row0 >= ptot[0]) return;          // uniform over block
  int e = 0;
  if (off) {
#pragma unroll
    for (int i = 0; i < E_; ++i)
      if (row0 >= off[i] && row0 < off[i] + pcnt[i]) e = i;
  }
  const __bf16* Wt = WtAll + (size_t)e * K * N;
  const float* bias = biasAll + (size_t)e * N;
  int n0 = blockIdx.x * 64;

  // staging map: tid -> row r (0..31, +32 second phase), 16B chunk sub (0..3)
  int r = tid >> 2, sub = tid & 3;
  int aR0 = row0 + r, aR1 = row0 + r + 32;
  if (gather) { aR0 = assign_tok[aR0]; aR1 = assign_tok[aR1]; }
  const __bf16* aSrc0 = A + (size_t)aR0 * K + sub * 8;
  const __bf16* aSrc1 = A + (size_t)aR1 * K + sub * 8;
  const __bf16* bSrc0 = Wt + (size_t)(n0 + r) * K + sub * 8;
  const __bf16* bSrc1 = Wt + (size_t)(n0 + r + 32) * K + sub * 8;
  unsigned la0 = (unsigned)(r * 32 + sub * 8);
  unsigned la1 = (unsigned)((r + 32) * 32 + sub * 8);

  int lane = tid & 31;
  int wv = tid >> 5;
  int col = lane & 15;

  auto stage = [&](int bi, int k0) {
    async_cp16(aSrc0 + k0, &sA[bi][la0]);
    async_cp16(aSrc1 + k0, &sA[bi][la1]);
    async_cp16(bSrc0 + k0, &sB[bi][la0]);
    async_cp16(bSrc1 + k0, &sB[bi][la1]);
  };

  v8f acc0 = {}, acc1 = {}, acc2 = {}, acc3 = {};
  int buf = 0;
  stage(0, 0);
  for (int k0 = 0; k0 < K; k0 += 32) {
    wait_async_all();                 // own async stores to LDS landed
    __syncthreads();                  // cross-wave visibility; prior reads done
    if (k0 + 32 < K) stage(buf ^ 1, k0 + 32);
    // hoist all 10 ds_load_b128 before the WMMAs so LDS latency overlaps
    v16bf a  = load_frag(&sA[buf][(wv * 16 + col) * 32], lane);
    v16bf b0 = load_frag(&sB[buf][(col) * 32], lane);
    v16bf b1 = load_frag(&sB[buf][(16 + col) * 32], lane);
    v16bf b2 = load_frag(&sB[buf][(32 + col) * 32], lane);
    v16bf b3 = load_frag(&sB[buf][(48 + col) * 32], lane);
    acc0 = wmma_bf16(a, b0, acc0);
    acc1 = wmma_bf16(a, b1, acc1);
    acc2 = wmma_bf16(a, b2, acc2);
    acc3 = wmma_bf16(a, b3, acc3);
    buf ^= 1;
  }

  int m0 = row0 + wv * 16;
  int rhalf = (lane & 16) ? 8 : 0;
  v8f accs[4] = {acc0, acc1, acc2, acc3};
#pragma unroll
  for (int ns = 0; ns < 4; ++ns) {
#pragma unroll
    for (int rr = 0; rr < 8; ++rr) {
      int row = m0 + rr + rhalf;
      int c = n0 + ns * 16 + col;
      float v = accs[ns][rr] + bias[c];
      if (gelu) v = 0.5f * v * (1.f + erff(v * 0.70710678118654752f));  // exact GELU
      size_t idx = (size_t)row * N + c;
      if (residual) v += residual[idx];
      if (outF) outF[idx] = v;
      if (outB) outB[idx] = f2bf(v);
    }
  }
}

// ---------------- RoPE + head layout (q,k) and V transpose -------------------
__global__ void rope_kernel(const float* __restrict__ qf, const float* __restrict__ kf,
                            const float* __restrict__ vf, __bf16* __restrict__ qh,
                            __bf16* __restrict__ kh, __bf16* __restrict__ vT) {
  int t = blockIdx.x;
  int b = t / S_, s = t - b * S_;
  for (int c = threadIdx.x; c < D_; c += blockDim.x) {
    int h = c >> 6, d = c & 63;
    int i = d & 31;
    float ang = (float)s * __expf(-(float)i * (9.210340371976184f / 32.f));
    float cs = __cosf(ang), sn = __sinf(ang);
    size_t src = (size_t)t * D_ + c;
    float qv = qf[src], kv = kf[src];
    float qp = (d < 32) ? -qf[src + 32] : qf[src - 32];
    float kp = (d < 32) ? -kf[src + 32] : kf[src - 32];
    size_t dsth = ((size_t)(b * H_ + h) * S_ + s) * HD_ + d;
    qh[dsth] = f2bf(qv * cs + qp * sn);
    kh[dsth] = f2bf(kv * cs + kp * sn);
    vT[((size_t)(b * H_ + h) * HD_ + d) * S_ + s] = f2bf(vf[src]);
  }
}

// ---------------- flash attention: 1 wave = 16 queries of one (b,h) ----------
__global__ __launch_bounds__(32) void attn_kernel(
    const __bf16* __restrict__ qh, const __bf16* __restrict__ kh,
    const __bf16* __restrict__ vT, __bf16* __restrict__ ctxB) {
  __shared__ __align__(16) __bf16 Pls[16 * 32];
  int lane = threadIdx.x;
  int qt = blockIdx.x, bh = blockIdx.y;
  int b = bh >> 4, h = bh & 15;
  int q0 = qt * 16;
  int col = lane & 15;
  int rhalf = (lane & 16) ? 8 : 0;

  const __bf16* qrow = qh + ((size_t)bh * S_ + q0 + col) * HD_;
  v16bf aQ0 = load_frag(qrow, lane);
  v16bf aQ1 = load_frag(qrow + 32, lane);

  float mrow[8], lrow[8];
  v8f zero = {};
  v8f ctx[4] = {zero, zero, zero, zero};
#pragma unroll
  for (int r = 0; r < 8; ++r) { mrow[r] = -3.0e38f; lrow[r] = 0.f; }

  int nKB = (q0 + 15) / 32 + 1;
  for (int kb = 0; kb < nKB; ++kb) {
    int k0 = kb * 32;
    const __bf16* krow = kh + ((size_t)bh * S_ + k0 + col) * HD_;
    v16bf bA0 = load_frag(krow, lane);
    v16bf bA1 = load_frag(krow + 32, lane);
    v16bf bB0 = load_frag(krow + (size_t)16 * HD_, lane);
    v16bf bB1 = load_frag(krow + (size_t)16 * HD_ + 32, lane);
    v8f sA = zero, sB = zero;
    sA = wmma_bf16(aQ0, bA0, sA);
    sA = wmma_bf16(aQ1, bA1, sA);
    sB = wmma_bf16(aQ0, bB0, sB);
    sB = wmma_bf16(aQ1, bB1, sB);
#pragma unroll
    for (int r = 0; r < 8; ++r) {
      int row = q0 + r + rhalf;
      float a = sA[r] * 0.125f;
      if (k0 + col > row) a = -1e30f;
      float bv = sB[r] * 0.125f;
      if (k0 + 16 + col > row) bv = -1e30f;
      float mx = fmaxf(a, bv);
#pragma unroll
      for (int off = 1; off < 16; off <<= 1) mx = fmaxf(mx, __shfl_xor(mx, off, 32));
      float mnew = fmaxf(mrow[r], mx);
      float alpha = __expf(mrow[r] - mnew);
      mrow[r] = mnew;
      float pa = __expf(a - mnew);
      float pb = __expf(bv - mnew);
      float sum = pa + pb;
#pragma unroll
      for (int off = 1; off < 16; off <<= 1) sum += __shfl_xor(sum, off, 32);
      lrow[r] = lrow[r] * alpha + sum;
#pragma unroll
      for (int ns = 0; ns < 4; ++ns) ctx[ns][r] = ctx[ns][r] * alpha;
      Pls[(r + rhalf) * 32 + col] = f2bf(pa);
      Pls[(r + rhalf) * 32 + 16 + col] = f2bf(pb);
    }
    __syncthreads();
    v16bf aP = load_frag(&Pls[(lane & 15) * 32], lane);
#pragma unroll
    for (int ns = 0; ns < 4; ++ns) {
      const __bf16* vrow = vT + ((size_t)bh * HD_ + ns * 16 + col) * S_ + k0;
      v16bf bV = load_frag(vrow, lane);
      ctx[ns] = wmma_bf16(aP, bV, ctx[ns]);
    }
    __syncthreads();
  }
#pragma unroll
  for (int r = 0; r < 8; ++r) {
    float inv = 1.f / lrow[r];
    int row = q0 + r + rhalf;
    size_t base = (size_t)(b * S_ + row) * D_ + h * HD_;
#pragma unroll
    for (int ns = 0; ns < 4; ++ns)
      ctxB[base + ns * 16 + col] = f2bf(ctx[ns][r] * inv);
  }
}

// ---------------- router / top-k / deterministic routing lists ---------------
__global__ void router_kernel(const float* __restrict__ xn2, const float* __restrict__ rw,
                              const float* __restrict__ rb, float* __restrict__ logits) {
  int t = blockIdx.x, e = threadIdx.x;
  if (e >= E_) return;
  const float* xr = xn2 + (size_t)t * D_;
  float s = rb[e];
  for (int d = 0; d < D_; ++d) s += xr[d] * rw[d * E_ + e];
  logits[t * E_ + e] = s;
}

__global__ void topk_kernel(const float* __restrict__ logits, int* __restrict__ topi,
                            float* __restrict__ topw) {
  int t = blockIdx.x * blockDim.x + threadIdx.x;
  if (t >= T_) return;
  float v[E_];
#pragma unroll
  for (int i = 0; i < E_; ++i) v[i] = logits[t * E_ + i];
  int i0 = 0;
#pragma unroll
  for (int i = 1; i < E_; ++i) if (v[i] > v[i0]) i0 = i;
  int i1 = (i0 == 0) ? 1 : 0;
#pragma unroll
  for (int i = 0; i < E_; ++i) if (i != i0 && v[i] > v[i1]) i1 = i;
  float e1 = __expf(v[i1] - v[i0]);
  float w0 = 1.f / (1.f + e1);
  topi[2 * t] = i0;  topi[2 * t + 1] = i1;
  topw[2 * t] = w0;  topw[2 * t + 1] = e1 * w0;
}

// Deterministic (no atomics) per-expert gather lists, padded to 64-row tiles.
__global__ void build_lists(const int* __restrict__ topi, int* __restrict__ assign_tok,
                            int* __restrict__ tok_row, int* __restrict__ off,
                            int* __restrict__ pcnt, int* __restrict__ ptot) {
  __shared__ int off_s[E_], pcnt_s[E_];
  int e = threadIdx.x;
  if (e < E_) {
    int c = 0;
    for (int t = 0; t < T_; ++t) {
      if (topi[2 * t] == e) c++;
      if (topi[2 * t + 1] == e) c++;
    }
    pcnt_s[e] = (c + 63) & ~63;       // pad to 64-row macro tiles
  }
  __syncthreads();
  if (e == 0) {
    int o = 0;
    for (int i = 0; i < E_; ++i) { off_s[i] = o; o += pcnt_s[i]; }
    ptot[0] = o;
  }
  __syncthreads();
  if (e < E_) {
    int r = off_s[e];
    for (int t = 0; t < T_; ++t) {
      if (topi[2 * t] == e)     { assign_tok[r] = t; tok_row[2 * t] = r;     r++; }
      if (topi[2 * t + 1] == e) { assign_tok[r] = t; tok_row[2 * t + 1] = r; r++; }
    }
    int end = off_s[e] + pcnt_s[e];
    for (; r < end; ++r) assign_tok[r] = 0;    // pad rows (never combined)
    off[e] = off_s[e];
    pcnt[e] = pcnt_s[e];
  }
}

// ---------------- final combine: out = xres + w0*y[r0] + w1*y[r1] ------------
__global__ void combine_kernel(const float* __restrict__ xres, const float* __restrict__ y_ws,
                               const int* __restrict__ tok_row, const float* __restrict__ topw,
                               float* __restrict__ out) {
  int t = blockIdx.x;
  int r0 = tok_row[2 * t], r1 = tok_row[2 * t + 1];
  float w0 = topw[2 * t], w1 = topw[2 * t + 1];
  for (int d = threadIdx.x; d < D_; d += blockDim.x) {
    out[(size_t)t * D_ + d] = xres[(size_t)t * D_ + d] +
                              w0 * y_ws[(size_t)r0 * D_ + d] +
                              w1 * y_ws[(size_t)r1 * D_ + d];
  }
}

// =============================== host side ===================================
extern "C" void kernel_launch(void* const* d_in, const int* in_sizes, int n_in,
                              void* d_out, int out_size, void* d_ws, size_t ws_size,
                              hipStream_t stream) {
  (void)in_sizes; (void)n_in; (void)out_size; (void)ws_size;
  const float* x           = (const float*)d_in[0];
  const float* attn_norm_w = (const float*)d_in[1];
  const float* ffn_norm_w  = (const float*)d_in[2];
  const float* wq = (const float*)d_in[3];  const float* bq = (const float*)d_in[4];
  const float* wk = (const float*)d_in[5];  const float* bk = (const float*)d_in[6];
  const float* wv = (const float*)d_in[7];  const float* bv = (const float*)d_in[8];
  const float* wo = (const float*)d_in[9];  const float* bo = (const float*)d_in[10];
  const float* router_w = (const float*)d_in[11]; const float* router_b = (const float*)d_in[12];
  const float* w1 = (const float*)d_in[13]; const float* b1 = (const float*)d_in[14];
  const float* w2 = (const float*)d_in[15]; const float* b2 = (const float*)d_in[16];

  float* out_x      = (float*)d_out;
  float* out_logits = out_x + (size_t)T_ * D_;

  // ---- carve workspace ----
  char* p = (char*)d_ws;
  auto carve = [&](size_t bytes) -> char* {
    char* r = p; p += (bytes + 255) & ~(size_t)255; return r;
  };
  __bf16* wqT  = (__bf16*)carve((size_t)D_ * D_ * 2);
  __bf16* wkT  = (__bf16*)carve((size_t)D_ * D_ * 2);
  __bf16* wvT  = (__bf16*)carve((size_t)D_ * D_ * 2);
  __bf16* woT  = (__bf16*)carve((size_t)D_ * D_ * 2);
  __bf16* w1T  = (__bf16*)carve((size_t)E_ * D_ * HID_ * 2);
  __bf16* w2T  = (__bf16*)carve((size_t)E_ * D_ * HID_ * 2);
  __bf16* xn_b = (__bf16*)carve((size_t)T_ * D_ * 2);
  float*  qf   = (float*)carve((size_t)T_ * D_ * 4);
  float*  kf   = (float*)carve((size_t)T_ * D_ * 4);
  float*  vf   = (float*)carve((size_t)T_ * D_ * 4);
  __bf16* qh   = (__bf16*)carve((size_t)T_ * D_ * 2);
  __bf16* kh   = (__bf16*)carve((size_t)T_ * D_ * 2);
  __bf16* vTb  = (__bf16*)carve((size_t)T_ * D_ * 2);
  __bf16* ctxB = (__bf16*)carve((size_t)T_ * D_ * 2);
  float*  xres = (float*)carve((size_t)T_ * D_ * 4);
  __bf16* xn2b = (__bf16*)carve((size_t)T_ * D_ * 2);
  float*  xn2f = (float*)carve((size_t)T_ * D_ * 4);
  int*   topi       = (int*)carve((size_t)T_ * 2 * 4);
  float* topw       = (float*)carve((size_t)T_ * 2 * 4);
  int*   tok_row    = (int*)carve((size_t)T_ * 2 * 4);
  int*   assign_tok = (int*)carve((size_t)MAXROWS * 4);
  int*   off_g      = (int*)carve(E_ * 4);
  int*   pcnt_g     = (int*)carve(E_ * 4);
  int*   ptot_g     = (int*)carve(4);
  __bf16* h_ws = (__bf16*)carve((size_t)MAXROWS * HID_ * 2);
  float*  y_ws = (float*)carve((size_t)MAXROWS * D_ * 4);

  dim3 b328(32, 8, 1);
  transposeW<<<dim3(32, 32, 1),   b328, 0, stream>>>(wq, wqT, D_, D_);
  transposeW<<<dim3(32, 32, 1),   b328, 0, stream>>>(wk, wkT, D_, D_);
  transposeW<<<dim3(32, 32, 1),   b328, 0, stream>>>(wv, wvT, D_, D_);
  transposeW<<<dim3(32, 32, 1),   b328, 0, stream>>>(wo, woT, D_, D_);
  transposeW<<<dim3(32, 128, E_), b328, 0, stream>>>(w1, w1T, D_, HID_);
  transposeW<<<dim3(128, 32, E_), b328, 0, stream>>>(w2, w2T, HID_, D_);

  rmsnorm_kernel<<<T_, 256, 0, stream>>>(x, attn_norm_w, xn_b, nullptr);

  // QKV projections: 64x64 macro tiles, async-LDS staged
  gemm_tiled<<<dim3(D_ / 64, T_ / 64), 128, 0, stream>>>(
      xn_b, wqT, bq, nullptr, qf, nullptr, D_, D_,
      nullptr, nullptr, nullptr, nullptr, 0, 0);
  gemm_tiled<<<dim3(D_ / 64, T_ / 64), 128, 0, stream>>>(
      xn_b, wkT, bk, nullptr, kf, nullptr, D_, D_,
      nullptr, nullptr, nullptr, nullptr, 0, 0);
  gemm_tiled<<<dim3(D_ / 64, T_ / 64), 128, 0, stream>>>(
      xn_b, wvT, bv, nullptr, vf, nullptr, D_, D_,
      nullptr, nullptr, nullptr, nullptr, 0, 0);

  rope_kernel<<<T_, 256, 0, stream>>>(qf, kf, vf, qh, kh, vTb);

  attn_kernel<<<dim3(S_ / 16, B_ * H_), 32, 0, stream>>>(qh, kh, vTb, ctxB);

  // output projection + residual
  gemm_tiled<<<dim3(D_ / 64, T_ / 64), 128, 0, stream>>>(
      ctxB, woT, bo, x, xres, nullptr, D_, D_,
      nullptr, nullptr, nullptr, nullptr, 0, 0);

  rmsnorm_kernel<<<T_, 256, 0, stream>>>(xres, ffn_norm_w, xn2b, xn2f);

  router_kernel<<<T_, 8, 0, stream>>>(xn2f, router_w, router_b, out_logits);
  topk_kernel<<<(T_ + 255) / 256, 256, 0, stream>>>(out_logits, topi, topw);
  build_lists<<<1, 32, 0, stream>>>(topi, assign_tok, tok_row, off_g, pcnt_g, ptot_g);

  // expert GEMMs: h = gelu(xn2@w1+b1)  then  y = h@w2+b2
  gemm_tiled<<<dim3(HID_ / 64, MAXROWS / 64), 128, 0, stream>>>(
      xn2b, w1T, b1, nullptr, nullptr, h_ws, D_, HID_,
      off_g, pcnt_g, ptot_g, assign_tok, 1, 1);
  gemm_tiled<<<dim3(D_ / 64, MAXROWS / 64), 128, 0, stream>>>(
      h_ws, w2T, b2, nullptr, y_ws, nullptr, HID_, D_,
      off_g, pcnt_g, ptot_g, assign_tok, 0, 0);

  combine_kernel<<<T_, 256, 0, stream>>>(xres, y_ws, tok_row, topw, out_x);
}
